// CAMILSelfAttention_14508399525960
// MI455X (gfx1250) — matrix operand order repaired and checked by reading, not compiled, not run on hardware
//
#include <hip/hip_runtime.h>
#include <hip/hip_bf16.h>
#include <math.h>

// ---------------------------------------------------------------------------
// CDNA5 (gfx1250) fused CAMIL self-attention.
// B=2, N=8192, IN_DIM=ATT_DIM=512.
// ---------------------------------------------------------------------------

typedef __attribute__((ext_vector_type(16))) __bf16        v16bf;
typedef __attribute__((ext_vector_type(8)))  float          v8f;
typedef __attribute__((ext_vector_type(8)))  unsigned int   v8u;

#define BATCH   2
#define NN      8192
#define DIM     512

// Pack two f32 into one dword of bf16 (lo -> [15:0], hi -> [31:16]).
// CDNA5 has bf16 VALU; use the packed convert (1 instruction for 2 elements)
// instead of the ~5-VALU-op manual RNE sequence.
__device__ __forceinline__ unsigned pack_bf16(float lo, float hi) {
#if __has_builtin(__builtin_amdgcn_cvt_pk_bf16_f32)
    typedef __attribute__((ext_vector_type(2))) __bf16 v2bf;
    v2bf r = __builtin_amdgcn_cvt_pk_bf16_f32(lo, hi);
    return __builtin_bit_cast(unsigned, r);
#else
    unsigned r;
    asm("v_cvt_pk_bf16_f32 %0, %1, %2" : "=v"(r) : "v"(lo), "v"(hi));
    return r;
#endif
}

__device__ __forceinline__ unsigned short f32_to_bf16(float f) {
    unsigned u = __builtin_bit_cast(unsigned, f);
    u += 0x7FFFu + ((u >> 16) & 1u);
    return (unsigned short)(u >> 16);
}

// Build a WMMA A/B fragment (16 bf16) from four float4s covering 16
// consecutive K values split as [base..base+7] and [base+8..base+15].
__device__ __forceinline__ v16bf make_frag(float4 lo0, float4 lo1,
                                           float4 hi0, float4 hi1) {
    v8u u;
    u[0] = pack_bf16(lo0.x, lo0.y);
    u[1] = pack_bf16(lo0.z, lo0.w);
    u[2] = pack_bf16(lo1.x, lo1.y);
    u[3] = pack_bf16(lo1.z, lo1.w);
    u[4] = pack_bf16(hi0.x, hi0.y);
    u[5] = pack_bf16(hi0.z, hi0.w);
    u[6] = pack_bf16(hi1.x, hi1.y);
    u[7] = pack_bf16(hi1.z, hi1.w);
    return __builtin_bit_cast(v16bf, u);
}

// ---------------------------------------------------------------------------
// Kernel 0: zero the score accumulator w[B*N]
// ---------------------------------------------------------------------------
__global__ void zero_w_kernel(float* __restrict__ w) {
    int i = blockIdx.x * blockDim.x + threadIdx.x;
    if (i < BATCH * NN) w[i] = 0.0f;
}

// ---------------------------------------------------------------------------
// Kernel 1: fused projections via WMMA bf16.
//   Y = X(16384x512) @ [Wqk | Wv] (512 x 1536)
//   cols [0,512)    -> q   (f32, row-major)                -> qf
//   cols [512,1024) -> k*mask (bf16, TRANSPOSED [b][d][j]) -> kT
//   cols [1024,1536)-> v   (f32, row-major)                -> out (d_out)
// Grid: (256, 96). blockIdx.y = column tile (scalar region select);
// 4 waves / block, one 16x16 tile per wave (4 row tiles per block).
// ---------------------------------------------------------------------------
__global__ void proj_kernel(const float* __restrict__ X,
                            const float* __restrict__ Wqk,
                            const float* __restrict__ Wv,
                            const float* __restrict__ mask,
                            float* __restrict__ qf,
                            unsigned short* __restrict__ kT,
                            float* __restrict__ vout) {
    const int lane = threadIdx.x & 31;
    const int wv   = threadIdx.x >> 5;
    const int nl   = lane & 15;
    const int half = lane >> 4;

    const int rowTile = blockIdx.x * 4 + wv;   // [0, 1024)
    const int c0      = blockIdx.y * 16;       // scalar: output column tile
    const int r0      = rowTile * 16;          // global row in [0, B*N)

    const float* xrow = X + (size_t)(r0 + nl) * DIM;   // A: row m = lane&15

    // scalar (SGPR) weight region select
    const float* W;
    int wstride, wc;
    if (c0 < 1024) { W = Wqk; wstride = 2 * DIM; wc = c0; }
    else           { W = Wv;  wstride = DIM;     wc = c0 - 1024; }

    v8f acc = {};
    for (int kk = 0; kk < DIM; kk += 32) {
        // ---- A tile: X[16 x 32] -> bf16 (lane = row, K = half*16+s) ----
        float4 a0 = *(const float4*)(xrow + kk + half * 16);
        float4 a1 = *(const float4*)(xrow + kk + half * 16 + 4);
        float4 a2 = *(const float4*)(xrow + kk + half * 16 + 8);
        float4 a3 = *(const float4*)(xrow + kk + half * 16 + 12);
        v16bf a = make_frag(a0, a1, a2, a3);

        // ---- B tile: W[32 x 16] -> bf16 (lane = col n, K = half*16+2p..) --
        v8u bu;
        #pragma unroll
        for (int p = 0; p < 8; ++p) {
            int kb = half * 16 + 2 * p;
            float b0 = W[(size_t)(kk + kb)     * wstride + wc + nl];
            float b1 = W[(size_t)(kk + kb + 1) * wstride + wc + nl];
            bu[p] = pack_bf16(b0, b1);
        }
        v16bf bb = __builtin_bit_cast(v16bf, bu);

        acc = __builtin_amdgcn_wmma_f32_16x16x32_bf16(false, a, false, bb,
                                                      (short)0, acc, false, false);
    }

    // ---- epilogue: C element (m = r + 8*half, n = lane&15) ----
    if (c0 < 512) {
        #pragma unroll
        for (int r = 0; r < 8; ++r) {
            const int row = r0 + r + half * 8;
            qf[(size_t)row * DIM + c0 + nl] = acc[r];
        }
    } else if (c0 < 1024) {
        const int d = (c0 - 512) + nl;
        #pragma unroll
        for (int r = 0; r < 8; ++r) {
            const int row = r0 + r + half * 8;
            const int b   = row >> 13;        // row / N
            const int jn  = row & (NN - 1);   // row % N
            float km = acc[r] * mask[row];
            kT[((size_t)b * DIM + d) * NN + jn] = f32_to_bf16(km);
        }
    } else {
        #pragma unroll
        for (int r = 0; r < 8; ++r) {
            const int row = r0 + r + half * 8;
            vout[(size_t)row * DIM + (c0 - 1024) + nl] = acc[r];
        }
    }
}

// ---------------------------------------------------------------------------
// Kernel 2: fused  w_i += inv_scale * q_i . (adj @ k_m)_i  via WMMA bf16.
// Grid: B * N/16 blocks (b, 16-row block). Block: 128 threads = 4 waves.
// Wave wv covers d-range [wv*128, wv*128+128): 8 accumulator tiles.
// j-loop over N in steps of 32 (K of the bf16 WMMA). adj read once from HBM.
// ---------------------------------------------------------------------------
__global__ void score_kernel(const float* __restrict__ adj,
                             const float* __restrict__ qf,
                             const unsigned short* __restrict__ kT,
                             float* __restrict__ w) {
    const int lane = threadIdx.x & 31;
    const int wvid = threadIdx.x >> 5;
    const int nl   = lane & 15;
    const int half = lane >> 4;

    const int b  = blockIdx.x >> 9;     // / (N/16)
    const int it = blockIdx.x & 511;
    const int i0 = it * 16;

    const float* arow = adj + ((size_t)b * NN + i0 + nl) * NN;  // this lane's adj row
    const unsigned short* kTb = kT + (size_t)b * DIM * NN;

    v8f acc[8];
    #pragma unroll
    for (int t = 0; t < 8; ++t) acc[t] = (v8f){};

    for (int j = 0; j < NN; j += 32) {
        // prefetch next adj chunk for this row (streams 537MB once from HBM)
        if (j + 256 < NN) __builtin_prefetch(arow + j + 256, 0, 1);

        // ---- A tile from adj (fp32 -> bf16, aligned float4 gathers) ----
        float4 a0 = *(const float4*)(arow + j + half * 16);
        float4 a1 = *(const float4*)(arow + j + half * 16 + 4);
        float4 a2 = *(const float4*)(arow + j + half * 16 + 8);
        float4 a3 = *(const float4*)(arow + j + half * 16 + 12);
        v16bf a = make_frag(a0, a1, a2, a3);

        // ---- 8 B tiles from transposed bf16 k: contiguous 32B per lane ----
        #pragma unroll
        for (int t = 0; t < 8; ++t) {
            const int d = wvid * 128 + t * 16 + nl;            // column of this lane
            const unsigned short* col = kTb + (size_t)d * NN + j + half * 16;
            v8u raw = *(const v8u*)col;                        // K = base..base+15, 32B aligned
            v16bf bb = __builtin_bit_cast(v16bf, raw);
            acc[t] = __builtin_amdgcn_wmma_f32_16x16x32_bf16(false, a, false, bb,
                                                             (short)0, acc[t],
                                                             false, false);
        }
    }

    // ---- epilogue: dot with q, reduce across the 16-lane half, atomicAdd ----
    const float inv_scale = 0.04419417382415922f;   // 1/sqrt(512)
    #pragma unroll
    for (int r = 0; r < 8; ++r) {
        const int m = r + half * 8;
        const float* qrow = qf + ((size_t)b * NN + i0 + m) * DIM;
        float sum = 0.0f;
        #pragma unroll
        for (int t = 0; t < 8; ++t) {
            const int d = wvid * 128 + t * 16 + nl;
            sum += acc[t][r] * qrow[d];
        }
        // xor-shuffle reduce over the 16 lanes of this half (masks < 16)
        sum += __shfl_xor(sum, 1, 32);
        sum += __shfl_xor(sum, 2, 32);
        sum += __shfl_xor(sum, 4, 32);
        sum += __shfl_xor(sum, 8, 32);
        if (nl == 0) {
            atomicAdd(&w[b * NN + i0 + m], inv_scale * sum);
        }
    }
}

// ---------------------------------------------------------------------------
// Kernel 3a: per-batch softmax over N with mask fold-in.  grid = B blocks.
// ---------------------------------------------------------------------------
__global__ void softmax_kernel(const float* __restrict__ w,
                               const float* __restrict__ mask,
                               float* __restrict__ p) {
    __shared__ float red[1024];
    const int b   = blockIdx.x;
    const int tid = threadIdx.x;

    float tv[NN / 1024];
    float lmax = -INFINITY;
    #pragma unroll
    for (int c = 0; c < NN / 1024; ++c) {
        const int i = tid + c * 1024;
        float t = w[b * NN + i] * mask[b * NN + i];
        tv[c] = t;
        lmax = fmaxf(lmax, t);
    }
    red[tid] = lmax; __syncthreads();
    for (int s = 512; s > 0; s >>= 1) {
        if (tid < s) red[tid] = fmaxf(red[tid], red[tid + s]);
        __syncthreads();
    }
    const float mx = red[0]; __syncthreads();

    float ev[NN / 1024];
    float lsum = 0.0f;
    #pragma unroll
    for (int c = 0; c < NN / 1024; ++c) {
        ev[c] = __expf(tv[c] - mx);
        lsum += ev[c];
    }
    red[tid] = lsum; __syncthreads();
    for (int s = 512; s > 0; s >>= 1) {
        if (tid < s) red[tid] += red[tid + s];
        __syncthreads();
    }
    const float inv = 1.0f / red[0];
    #pragma unroll
    for (int c = 0; c < NN / 1024; ++c) {
        p[b * NN + tid + c * 1024] = ev[c] * inv;
    }
}

// ---------------------------------------------------------------------------
// Kernel 3b: out[b,i,:] *= p[b,i]   (out already holds v), float4 vectorized.
// ---------------------------------------------------------------------------
__global__ void scale_kernel(float* __restrict__ out, const float* __restrict__ p) {
    const size_t total4 = (size_t)BATCH * NN * DIM / 4;
    const size_t stride = (size_t)gridDim.x * blockDim.x;
    for (size_t idx = (size_t)blockIdx.x * blockDim.x + threadIdx.x;
         idx < total4; idx += stride) {
        const size_t row = idx / (DIM / 4);    // b*N + i
        float4 v = ((float4*)out)[idx];
        const float s = p[row];
        v.x *= s; v.y *= s; v.z *= s; v.w *= s;
        ((float4*)out)[idx] = v;
    }
}

// ---------------------------------------------------------------------------
// Host-side launcher (graph-capture safe: only kernel launches on `stream`).
// ---------------------------------------------------------------------------
extern "C" void kernel_launch(void* const* d_in, const int* in_sizes, int n_in,
                              void* d_out, int out_size, void* d_ws, size_t ws_size,
                              hipStream_t stream) {
    (void)in_sizes; (void)n_in; (void)out_size; (void)ws_size;

    const float* X    = (const float*)d_in[0];   // (B,N,512)
    const float* adj  = (const float*)d_in[1];   // (B,N,N)
    const float* mask = (const float*)d_in[2];   // (B,N)
    const float* Wqk  = (const float*)d_in[3];   // (512,1024)
    const float* Wv   = (const float*)d_in[4];   // (512,512)
    float*       out  = (float*)d_out;           // (B,N,512)

    char* ws = (char*)d_ws;
    float*          qf = (float*)ws;                              // 33,554,432 B
    unsigned short* kT = (unsigned short*)(ws + 33554432);        // 16,777,216 B
    float*          w  = (float*)(ws + 50331648);                 //     65,536 B
    float*          p  = (float*)(ws + 50397184);                 //     65,536 B

    zero_w_kernel<<<(BATCH * NN + 255) / 256, 256, 0, stream>>>(w);

    // grid (256 row-tile groups, 96 column tiles); 4 waves per block
    proj_kernel<<<dim3(256, 96), 128, 0, stream>>>(X, Wqk, Wv, mask, qf, kT, out);

    // B * N/16 = 1024 blocks, 4 waves each covering 128 of the 512 d-columns
    score_kernel<<<BATCH * (NN / 16), 128, 0, stream>>>(adj, qf, kT, w);

    softmax_kernel<<<BATCH, 1024, 0, stream>>>(w, mask, p);

    scale_kernel<<<4096, 256, 0, stream>>>(out, p);
}
